// GroupedQueryAttention_88527865905270
// MI455X (gfx1250) — matrix operand order repaired and checked by reading, not compiled, not run on hardware
//
#include <hip/hip_runtime.h>
#include <hip/hip_bf16.h>

// ---------------------------------------------------------------------------
// GroupedQueryAttention for MI455X (gfx1250): bf16 WMMA everywhere.
// B=4, T=2048, D=1024, H=16, G=4, hd=64.
// ---------------------------------------------------------------------------

typedef __attribute__((ext_vector_type(16))) __bf16 v16bf;
typedef __attribute__((ext_vector_type(8)))  __bf16 v8bf;
typedef __attribute__((ext_vector_type(8)))  float  v8f;

union U16bf { v16bf v; v8bf h[2]; };

static __device__ __forceinline__ v8f wmma_bf16(v16bf a, v16bf b, v8f c) {
  // D = A(16x32 bf16) * B(32x16 bf16) + C(16x16 f32)
  return __builtin_amdgcn_wmma_f32_16x16x32_bf16(
      /*neg_a=*/false, a, /*neg_b=*/false, b,
      /*c_mod=*/(short)0, c, /*reuse_a=*/false, /*reuse_b=*/false);
}

// A-matrix fragment, 16-bit A 16x32 (M x K), A row-major with leading dim lda.
// lanes 0-15 : row m0+l, K = {k0..k0+7, k0+16..k0+23}
// lanes 16-31: row m0+l, K = {k0+8..k0+15, k0+24..k0+31}
static __device__ __forceinline__ v16bf load_a_frag(const __bf16* A, int lda,
                                                    int m0, int k0, int lane) {
  const int l  = lane & 15;
  const int hi = lane >> 4;
  const __bf16* p = A + (size_t)(m0 + l) * lda + k0 + hi * 8;
  U16bf u;
  u.h[0] = *(const v8bf*)(p);
  u.h[1] = *(const v8bf*)(p + 16);
  return u.v;
}

// B-matrix fragment, 16-bit B 32x16 (K x N).  B supplied as W (N x K) row-major
// so B(k,n) = W[n*ldb + k]; per-lane K order matches load_a_frag.
// lanes 0-15 : n = n0+l, K = {k0..k0+7, k0+16..k0+23} ; lanes 16-31: n = n0+l, K = {k0+8..k0+15, k0+24..k0+31}
static __device__ __forceinline__ v16bf load_b_frag(const __bf16* W, int ldb,
                                                    int n0, int k0, int lane) {
  const int l  = lane & 15;
  const int hi = lane >> 4;
  const __bf16* p = W + (size_t)(n0 + l) * ldb + k0 + hi * 8;
  U16bf u;
  u.h[0] = *(const v8bf*)(p);
  u.h[1] = *(const v8bf*)(p + 16);
  return u.v;
}

// ---------------------------------------------------------------------------
// f32 -> bf16 conversion (grid-stride)
// ---------------------------------------------------------------------------
__global__ void cvt_f32_bf16(const float* __restrict__ in,
                             __bf16* __restrict__ out, int n) {
  int i = blockIdx.x * blockDim.x + threadIdx.x;
  const int stride = gridDim.x * blockDim.x;
  for (; i < n; i += stride) out[i] = (__bf16)in[i];
}

// ---------------------------------------------------------------------------
// C(MxN,f32) = A(MxK,bf16 rowmajor) @ W(NxK,bf16 rowmajor)^T
// block = 256 threads = 8 waves (2x4), wave tile 64x64 (16 WMMAs / K-step)
// ---------------------------------------------------------------------------
__global__ __launch_bounds__(256) void gemm_bt_bf16_f32(
    const __bf16* __restrict__ A, const __bf16* __restrict__ W,
    float* __restrict__ C, int M, int N, int K) {
  const int tid  = threadIdx.x;
  const int lane = tid & 31;
  const int wv   = tid >> 5;
  const int wm   = wv >> 2;          // 0..1
  const int wn   = wv & 3;           // 0..3
  const int m0   = blockIdx.x * 128 + wm * 64;
  const int n0   = blockIdx.y * 256 + wn * 64;
  const int l15  = lane & 15;

  v8f acc[4][4] = {};
  for (int k0 = 0; k0 < K; k0 += 32) {
    if (k0 + 32 < K) {   // pull next K-slab toward the caches
      __builtin_prefetch(A + (size_t)(m0 + l15) * K + k0 + 32, 0, 0);
      __builtin_prefetch(W + (size_t)(n0 + l15) * K + k0 + 32, 0, 0);
    }
    v16bf a[4], b[4];
#pragma unroll
    for (int i = 0; i < 4; ++i) a[i] = load_a_frag(A, K, m0 + 16 * i, k0, lane);
#pragma unroll
    for (int j = 0; j < 4; ++j) b[j] = load_b_frag(W, K, n0 + 16 * j, k0, lane);
#pragma unroll
    for (int i = 0; i < 4; ++i)
#pragma unroll
      for (int j = 0; j < 4; ++j)
        acc[i][j] = wmma_bf16(a[i], b[j], acc[i][j]);
  }

  const int hi = lane >> 4;
#pragma unroll
  for (int i = 0; i < 4; ++i)
#pragma unroll
    for (int j = 0; j < 4; ++j)
#pragma unroll
      for (int r = 0; r < 8; ++r)
        C[(size_t)(m0 + 16 * i + r + hi * 8) * N + n0 + 16 * j + l15] =
            acc[i][j][r];
}

// ---------------------------------------------------------------------------
// Fused RMS-norm (per 64-elem head) + interleaved RoPE, f32 in -> bf16 out.
// One wave per (token, head); lane = rotation pair index (hd/2 = 32 = wave32).
// ---------------------------------------------------------------------------
__global__ __launch_bounds__(256) void rmsnorm_rope(
    const float* __restrict__ src, __bf16* __restrict__ dst,
    const float* __restrict__ w, int heads, int T) {
  const int gtid = blockIdx.x * blockDim.x + threadIdx.x;
  const int wid  = gtid >> 5;
  const int lane = gtid & 31;
  const int rowlen = heads * 64;
  const int bt = wid / heads;
  const int h  = wid - bt * heads;
  const int t  = bt % T;

  const float* p = src + (size_t)bt * rowlen + h * 64;
  float x1 = p[2 * lane];
  float x2 = p[2 * lane + 1];

  float ss = x1 * x1 + x2 * x2;
#pragma unroll
  for (int m = 16; m >= 1; m >>= 1) ss += __shfl_xor(ss, m, 32);
  const float inv = rsqrtf(ss * (1.0f / 64.0f) + 1e-6f);

  const float n1 = x1 * inv * w[2 * lane];
  const float n2 = x2 * inv * w[2 * lane + 1];

  const float freq = powf(10000.0f, -(float)lane * (1.0f / 32.0f));
  const float ang  = (float)t * freq;
  const float c = cosf(ang);
  const float s = sinf(ang);

  __bf16* q = dst + (size_t)bt * rowlen + h * 64;
  q[2 * lane]     = (__bf16)(n1 * c - n2 * s);
  q[2 * lane + 1] = (__bf16)(n2 * c + n1 * s);
}

// ---------------------------------------------------------------------------
// Flash-style causal GQA attention, all matmuls via WMMA bf16.
// block = 128 threads = 4 waves; block handles one (b, h, 64-row q tile);
// wave w owns S/O rows 16w..16w+15 as WMMA C fragments.
// ---------------------------------------------------------------------------
__global__ __launch_bounds__(128) void attn_flash(
    const __bf16* __restrict__ qb, const __bf16* __restrict__ kb,
    const __bf16* __restrict__ vb, __bf16* __restrict__ ob, int T) {
  __shared__ __align__(16) __bf16 Pl[64 * 72];   // P tile (bf16, padded rows)
  __shared__ __align__(16) __bf16 Vt[64 * 72];   // V tile transposed [d][j]

  const int tid  = threadIdx.x;
  const int lane = tid & 31;
  const int wv   = tid >> 5;          // 0..3 -> rows 16w..16w+15
  const int l15  = lane & 15;
  const int hi   = lane >> 4;

  const int qt = blockIdx.x;          // q tile (64 rows)
  const int h  = blockIdx.y;          // head
  const int b  = blockIdx.z;          // batch
  const int g  = h >> 2;              // kv group
  const int D  = 1024, Dk = 256;

  const size_t qrow0 = (size_t)(b * T + qt * 64);

  // Q A-fragments for this wave's 16 rows (head-dim K = 64 -> 2 K-steps)
  const __bf16* qbase = qb + qrow0 * D + h * 64;
  const v16bf qa0 = load_a_frag(qbase, D, 16 * wv, 0, lane);
  const v16bf qa1 = load_a_frag(qbase, D, 16 * wv, 32, lane);

  v8f accO[4] = {};
  float m_st[8], l_st[8];
#pragma unroll
  for (int r = 0; r < 8; ++r) { m_st[r] = -3.0e38f; l_st[r] = 0.0f; }

  for (int kt = 0; kt <= qt; ++kt) {
    const size_t kvrow0 = (size_t)(b * T + kt * 64);

    // ---- stage V tile transposed into LDS (cooperative) ----
    __syncthreads();
    for (int e = tid; e < 64 * 64; e += 128) {
      const int j = e >> 6;       // kv index within tile
      const int d = e & 63;       // head-dim index
      Vt[d * 72 + j] = vb[(kvrow0 + j) * Dk + g * 64 + d];
    }
    __syncthreads();

    // ---- S = Q @ K^T  (K fragments straight from global, contiguous) ----
    v8f accS[4] = {};
    const __bf16* kbase = kb + kvrow0 * Dk + g * 64;
#pragma unroll
    for (int ks = 0; ks < 2; ++ks) {
      const v16bf qa = ks ? qa1 : qa0;
#pragma unroll
      for (int nt = 0; nt < 4; ++nt) {
        const v16bf bf = load_b_frag(kbase, Dk, 16 * nt, 32 * ks, lane);
        accS[nt] = wmma_bf16(qa, bf, accS[nt]);
      }
    }

    // ---- online softmax update, per C-fragment row ----
#pragma unroll
    for (int r = 0; r < 8; ++r) {
      const int qi = qt * 64 + 16 * wv + r + hi * 8;
      float sv[4];
#pragma unroll
      for (int nt = 0; nt < 4; ++nt) {
        float s = accS[nt][r] * 0.125f;             // 1/sqrt(64)
        const int kj = kt * 64 + 16 * nt + l15;
        if (kj > qi) s = -3.0e38f;                  // causal mask
        sv[nt] = s;
      }
      float mx = fmaxf(fmaxf(sv[0], sv[1]), fmaxf(sv[2], sv[3]));
#pragma unroll
      for (int m = 1; m <= 8; m <<= 1) mx = fmaxf(mx, __shfl_xor(mx, m, 32));
      const float m_new = fmaxf(m_st[r], mx);
      const float scale = __expf(m_st[r] - m_new);
      float rsum = 0.0f;
#pragma unroll
      for (int nt = 0; nt < 4; ++nt) {
        const float pv = __expf(sv[nt] - m_new);
        rsum += pv;
        Pl[(16 * wv + r + hi * 8) * 72 + 16 * nt + l15] = (__bf16)pv;
      }
#pragma unroll
      for (int m = 1; m <= 8; m <<= 1) rsum += __shfl_xor(rsum, m, 32);
      l_st[r] = l_st[r] * scale + rsum;
      m_st[r] = m_new;
#pragma unroll
      for (int nt = 0; nt < 4; ++nt) accO[nt][r] *= scale;
    }

    // ---- O += P @ V  (P rows are this wave's own LDS rows: no barrier) ----
#pragma unroll
    for (int ks = 0; ks < 2; ++ks) {
      const v16bf pa = load_a_frag(Pl, 72, 16 * wv, 32 * ks, lane);
#pragma unroll
      for (int nt = 0; nt < 4; ++nt) {
        const v16bf bf = load_b_frag(Vt, 72, 16 * nt, 32 * ks, lane);
        accO[nt] = wmma_bf16(pa, bf, accO[nt]);
      }
    }
  }

  // ---- normalize + store (bf16, feeds output-projection WMMA GEMM) ----
#pragma unroll
  for (int r = 0; r < 8; ++r) {
    const float invl = 1.0f / l_st[r];
    const size_t row = qrow0 + 16 * wv + r + hi * 8;
#pragma unroll
    for (int nt = 0; nt < 4; ++nt)
      ob[row * 1024 + h * 64 + 16 * nt + l15] = (__bf16)(accO[nt][r] * invl);
  }
}

// ---------------------------------------------------------------------------
// Launch
// ---------------------------------------------------------------------------
extern "C" void kernel_launch(void* const* d_in, const int* in_sizes, int n_in,
                              void* d_out, int out_size, void* d_ws,
                              size_t ws_size, hipStream_t stream) {
  (void)in_sizes; (void)n_in; (void)out_size; (void)ws_size;
  constexpr int Bb = 4, Tt = 2048, Dd = 1024, Hh = 16, Gg = 4;
  constexpr int BT = Bb * Tt;                // 8192
  constexpr int Dk = Gg * 64;                // 256

  const float* x   = (const float*)d_in[0];
  const float* Wq  = (const float*)d_in[1];
  const float* Wk  = (const float*)d_in[2];
  const float* Wv  = (const float*)d_in[3];
  const float* Wo  = (const float*)d_in[4];
  const float* qnw = (const float*)d_in[5];
  const float* knw = (const float*)d_in[6];
  float* out = (float*)d_out;

  char* ws = (char*)d_ws;
  size_t off = 0;
  auto take = [&](size_t bytes) -> void* {
    void* p = ws + off;
    off += (bytes + 255) & ~(size_t)255;
    return p;
  };

  __bf16* xb  = (__bf16*)take((size_t)BT * Dd * 2);
  __bf16* Wqb = (__bf16*)take((size_t)Dd * Dd * 2);
  __bf16* Wkb = (__bf16*)take((size_t)Dk * Dd * 2);
  __bf16* Wvb = (__bf16*)take((size_t)Dk * Dd * 2);
  __bf16* Wob = (__bf16*)take((size_t)Dd * Dd * 2);
  float*  qf  = (float*) take((size_t)BT * Dd * 4);
  float*  kf  = (float*) take((size_t)BT * Dk * 4);
  float*  vf  = (float*) take((size_t)BT * Dk * 4);
  __bf16* qbn = (__bf16*)take((size_t)BT * Dd * 2);
  __bf16* kbn = (__bf16*)take((size_t)BT * Dk * 2);
  __bf16* vbn = (__bf16*)take((size_t)BT * Dk * 2);
  __bf16* obf = (__bf16*)take((size_t)BT * Dd * 2);

  // 1) f32 -> bf16 for activations + weights
  cvt_f32_bf16<<<2048, 256, 0, stream>>>(x,  xb,  BT * Dd);
  cvt_f32_bf16<<<1024, 256, 0, stream>>>(Wq, Wqb, Dd * Dd);
  cvt_f32_bf16<<<512,  256, 0, stream>>>(Wk, Wkb, Dk * Dd);
  cvt_f32_bf16<<<512,  256, 0, stream>>>(Wv, Wvb, Dk * Dd);
  cvt_f32_bf16<<<1024, 256, 0, stream>>>(Wo, Wob, Dd * Dd);

  // 2) QKV projections (WMMA bf16 -> f32)
  gemm_bt_bf16_f32<<<dim3(BT / 128, Dd / 256), 256, 0, stream>>>(
      xb, Wqb, qf, BT, Dd, Dd);
  gemm_bt_bf16_f32<<<dim3(BT / 128, Dk / 256), 256, 0, stream>>>(
      xb, Wkb, kf, BT, Dk, Dd);
  gemm_bt_bf16_f32<<<dim3(BT / 128, Dk / 256), 256, 0, stream>>>(
      xb, Wvb, vf, BT, Dk, Dd);

  // 3) RMS norm + RoPE (f32 -> bf16); 4) v convert
  rmsnorm_rope<<<(BT * Hh) / 8, 256, 0, stream>>>(qf, qbn, qnw, Hh, Tt);
  rmsnorm_rope<<<(BT * Gg) / 8, 256, 0, stream>>>(kf, kbn, knw, Gg, Tt);
  cvt_f32_bf16<<<1024, 256, 0, stream>>>(vf, vbn, BT * Dk);

  // 5) causal GQA flash attention (WMMA bf16)
  attn_flash<<<dim3(Tt / 64, Hh, Bb), 128, 0, stream>>>(qbn, kbn, vbn, obf, Tt);

  // 6) output projection (WMMA bf16 -> f32 into d_out)
  gemm_bt_bf16_f32<<<dim3(BT / 128, Dd / 256), 256, 0, stream>>>(
      obf, Wob, out, BT, Dd, Dd);
}
